// xLSTMEncoderV2_45921790328944
// MI455X (gfx1250) — matrix operand order repaired and checked by reading, not compiled, not run on hardware
//
#include <hip/hip_runtime.h>

typedef __attribute__((ext_vector_type(16))) __bf16 v16bf;
typedef __attribute__((ext_vector_type(8)))  float  v8f;

#define SS   512
#define INC  128
#define DD   64
#define FD   256
#define ROWS 8          // batch rows per workgroup
#define NTHR 256        // 8 wave32

// ------------------- LDS layout (bytes) -------------------
#define OFF_CM    0                                   // [ROWS][DD][DD] f32 C_m (also setup staging)
#define OFF_WCT   (OFF_CM   + ROWS*DD*DD*4)           // WcT  bf16 [FD][INC]  (W_in@W_s)^T
#define OFF_RST   (OFF_WCT  + FD*INC*2)               // RsT  bf16 [FD][DD]
#define OFF_WQT   (OFF_RST  + FD*DD*2)                // [Wq|Wk*s|Wv|Wo]^T bf16 [FD][DD]
#define OFF_WFT   (OFF_WQT  + FD*DD*2)                // WfuseT bf16 [DD][INC]
#define OFF_G     (OFF_WFT  + DD*INC*2)               // G f32 [16][FD] (shared gemm out buffer)
#define OFF_XT    (OFF_G    + 16*FD*4)                // Xt  bf16 [16][INC]
#define OFF_HS    (OFF_XT   + 16*INC*2)               // Hs  bf16 [16][DD]
#define OFF_HSO   (OFF_HS   + 16*DD*2)                // Hso bf16 [16][DD]
#define OFF_CAT   (OFF_HSO  + 16*DD*2)                // [Hso|Hm] bf16 [16][INC]
#define OFF_CS    (OFF_CAT  + 16*INC*2)               // c_s f32 [ROWS][DD]
#define OFF_NS    (OFF_CS   + ROWS*DD*4)              // n_s
#define OFF_NM    (OFF_NS   + ROWS*DD*4)              // n_m
#define OFF_NUM   (OFF_NM   + ROWS*DD*4)              // num = C_m q
#define OFF_HM    (OFF_NUM  + ROWS*DD*4)              // h_m f32
#define OFF_BEFF  (OFF_HM   + ROWS*DD*4)              // b_eff f32 [FD]
#define OFF_VEC   (OFF_BEFF + FD*4)                   // [bo|b_fuse|wi|wf|ln_g|ln_b] f32 [6][DD]
#define OFF_SC    (OFF_VEC  + 6*DD*4)                 // imv[8] fmv[8] den[8]
#define SMEM_BYTES (OFF_SC + 3*ROWS*4)                // = 320096 B <= 320KB WGP LDS

#define WMMA_BF16(a,b,c) \
  __builtin_amdgcn_wmma_f32_16x16x32_bf16(false,(a),false,(b),(short)0,(c),false,false)

// A fragment: 16x32 bf16 tile, row-major source, ld in elements.
// lanes 0-15: M=lane, K in {0..7, 16..23}; lanes 16-31: M=lane-16, K in {8..15, 24..31}
__device__ __forceinline__ v16bf load_a(const __bf16* tile, int ld, int lane) {
  const int r = lane & 15;
  const __bf16* p = tile + r * ld + ((lane >> 4) << 3);
  v16bf v;
#pragma unroll
  for (int e = 0; e < 8; ++e) { v[e] = p[e]; v[e + 8] = p[e + 16]; }
  return v;
}
// B fragment: 32x16 bf16 tile from N-major (transposed) weights, row = N, ld = K stride.
// lanes 0-15: N=lane, K in {0..7, 16..23} ; lanes 16-31: N=lane-16, K in {8..15, 24..31}
__device__ __forceinline__ v16bf load_b(const __bf16* wt, int ld, int lane) {
  const int n = lane & 15;
  const __bf16* p = wt + n * ld + ((lane >> 4) << 3);
  v16bf v;
#pragma unroll
  for (int e = 0; e < 8; ++e) { v[e] = p[e]; v[e + 8] = p[e + 16]; }
  return v;
}
// D tile store: lanes 0-15 -> N=lane, M=vgpr ; lanes 16-31 -> N=lane-16, M=vgpr+8
__device__ __forceinline__ void store_d(float* out, int ld, int lane, v8f acc) {
  const int n = lane & 15, mb = (lane >> 4) << 3;
#pragma unroll
  for (int m = 0; m < 8; ++m) out[(mb + m) * ld + n] = acc[m];
}

__device__ __forceinline__ float wave_sum(float v) {
#pragma unroll
  for (int m = 16; m >= 1; m >>= 1) v += __shfl_xor(v, m, 32);
  return v;
}
__device__ __forceinline__ float sigmoidf(float x) { return 1.f / (1.f + __expf(-x)); }

__global__ void __launch_bounds__(NTHR, 1)
xlstm_fused_kernel(const float* __restrict__ x,     const float* __restrict__ W_in,
                   const float* __restrict__ b_in,  const float* __restrict__ W_s,
                   const float* __restrict__ R_s,   const float* __restrict__ b_s,
                   const float* __restrict__ ln_g,  const float* __restrict__ ln_b,
                   const float* __restrict__ Wq,    const float* __restrict__ Wk,
                   const float* __restrict__ Wv,    const float* __restrict__ wi,
                   const float* __restrict__ bip,   const float* __restrict__ wf,
                   const float* __restrict__ bfp,   const float* __restrict__ Wo,
                   const float* __restrict__ bo,    const float* __restrict__ W_fuse,
                   const float* __restrict__ b_fuse, float* __restrict__ out)
{
  extern __shared__ char smem[];
  const int tid  = threadIdx.x;
  const int lane = tid & 31;
  const int wv   = tid >> 5;
  const int b0   = blockIdx.x * ROWS;

  float*  Cm   = (float*)(smem + OFF_CM);
  __bf16* WcT  = (__bf16*)(smem + OFF_WCT);
  __bf16* RsT  = (__bf16*)(smem + OFF_RST);
  __bf16* WqT  = (__bf16*)(smem + OFF_WQT);
  __bf16* WfT  = (__bf16*)(smem + OFF_WFT);
  float*  G    = (float*)(smem + OFF_G);
  __bf16* Xt   = (__bf16*)(smem + OFF_XT);
  __bf16* Hs   = (__bf16*)(smem + OFF_HS);
  __bf16* Hso  = (__bf16*)(smem + OFF_HSO);
  __bf16* Cat  = (__bf16*)(smem + OFF_CAT);
  float*  cs   = (float*)(smem + OFF_CS);
  float*  ns   = (float*)(smem + OFF_NS);
  float*  nm   = (float*)(smem + OFF_NM);
  float*  numb = (float*)(smem + OFF_NUM);
  float*  hmb  = (float*)(smem + OFF_HM);
  float*  beff = (float*)(smem + OFF_BEFF);
  float*  vec  = (float*)(smem + OFF_VEC);   // [0]=bo [1]=b_fuse [2]=wi [3]=wf [4]=ln_g [5]=ln_b
  float*  sc   = (float*)(smem + OFF_SC);    // [0..7]=im [8..15]=fm [16..23]=den

  // ---------------- setup: stage W_in,W_s in the (not yet live) C_m region ----------------
  {
    float* sWin = Cm;             // [INC][DD]
    float* sWs  = Cm + INC * DD;  // [DD][FD]
    for (int i = tid; i < INC * DD; i += NTHR) sWin[i] = W_in[i];
    for (int i = tid; i < DD * FD; i += NTHR)  sWs[i]  = W_s[i];
    __syncthreads();
    // WcT[n][a] = sum_d W_in[a][d]*W_s[d][n] ; b_eff[n] = b_s[n] + sum_d b_in[d]*W_s[d][n]
    for (int i = tid; i < FD * INC; i += NTHR) {
      const int n = i >> 7, a = i & 127;
      float acc = 0.f;
      for (int d = 0; d < DD; ++d) acc += sWin[a * DD + d] * sWs[d * FD + n];
      WcT[n * INC + a] = (__bf16)acc;
    }
    for (int n = tid; n < FD; n += NTHR) {
      float acc = b_s[n];
      for (int d = 0; d < DD; ++d) acc += b_in[d] * sWs[d * FD + n];
      beff[n] = acc;
    }
    // other weights (independent LDS regions)
    for (int i = tid; i < FD * DD; i += NTHR) {      // RsT[n][d] = R_s[d][n]
      const int n = i >> 6, d = i & 63;
      RsT[n * DD + d] = (__bf16)R_s[d * FD + n];
    }
    for (int i = tid; i < FD * DD; i += NTHR) {      // [Wq|Wk*inv_sqrt_d|Wv|Wo]^T
      const int ng = i >> 6, d = i & 63;
      const int g = ng >> 6, n = ng & 63;
      const float* W = (g == 0) ? Wq : (g == 1) ? Wk : (g == 2) ? Wv : Wo;
      float v = W[d * DD + n];
      if (g == 1) v *= 0.125f;                       // 1/sqrt(64)
      WqT[ng * DD + d] = (__bf16)v;
    }
    for (int i = tid; i < DD * INC; i += NTHR) {     // WfT[n][k] = W_fuse[k][n]
      const int n = i >> 7, k = i & 127;
      WfT[n * INC + k] = (__bf16)W_fuse[k * DD + n];
    }
    if (tid < DD) {
      vec[0 * DD + tid] = bo[tid];     vec[1 * DD + tid] = b_fuse[tid];
      vec[2 * DD + tid] = wi[tid];     vec[3 * DD + tid] = wf[tid];
      vec[4 * DD + tid] = ln_g[tid];   vec[5 * DD + tid] = ln_b[tid];
    }
    __syncthreads();  // staging region now dead -> becomes C_m
    for (int i = tid; i < ROWS * DD * DD; i += NTHR) Cm[i] = 0.f;
    for (int i = tid; i < ROWS * DD;      i += NTHR) { cs[i] = 0.f; ns[i] = 1.f; nm[i] = 0.f; }
    for (int i = tid; i < 16 * INC; i += NTHR) { Xt[i] = (__bf16)0.f; Cat[i] = (__bf16)0.f; }
    for (int i = tid; i < 16 * DD;  i += NTHR) { Hs[i] = (__bf16)0.f; Hso[i] = (__bf16)0.f; }
  }
  __syncthreads();

  const float biS = bip[0], bfS = bfp[0];

  // =============================== recurrent scan ===============================
  for (int t = 0; t < SS; ++t) {
    // ---- stream x_t tile [8][128] -> bf16 LDS (coalesced float4) + prefetch t+1 ----
    {
      const int r = tid >> 5, c4 = (tid & 31) << 2;
      const float4 xv = *(const float4*)(x + ((size_t)(b0 + r) * SS + t) * INC + c4);
      Xt[r * INC + c4 + 0] = (__bf16)xv.x;  Xt[r * INC + c4 + 1] = (__bf16)xv.y;
      Xt[r * INC + c4 + 2] = (__bf16)xv.z;  Xt[r * INC + c4 + 3] = (__bf16)xv.w;
      if (t + 1 < SS)
        __builtin_prefetch(x + ((size_t)(b0 + r) * SS + (t + 1)) * INC + c4, 0, 0);
    }
    __syncthreads();

    // ---- GEMM1: G[16,256] = Xt@Wc + Hs@R_s  (bf16 WMMA, f32 accum) ----
    {
      v16bf ax[4], ah[2];
#pragma unroll
      for (int kk = 0; kk < 4; ++kk) ax[kk] = load_a(Xt + kk * 32, INC, lane);
#pragma unroll
      for (int kk = 0; kk < 2; ++kk) ah[kk] = load_a(Hs + kk * 32, DD, lane);
#pragma unroll
      for (int it = 0; it < 2; ++it) {
        const int n0 = (wv + it * 8) * 16;
        v8f acc = {0.f, 0.f, 0.f, 0.f, 0.f, 0.f, 0.f, 0.f};
#pragma unroll
        for (int kk = 0; kk < 4; ++kk)
          acc = WMMA_BF16(ax[kk], load_b(WcT + n0 * INC + kk * 32, INC, lane), acc);
#pragma unroll
        for (int kk = 0; kk < 2; ++kk)
          acc = WMMA_BF16(ah[kk], load_b(RsT + n0 * DD + kk * 32, DD, lane), acc);
        store_d(G + n0, FD, lane, acc);
      }
    }
    __syncthreads();

    // ---- sLSTM cell + LayerNorm (wave wv owns batch row wv) ----
    {
      const int r = wv;
      const float* Gr = G + r * FD;
      float hsv[2];
      float s1 = 0.f, s2 = 0.f;
#pragma unroll
      for (int u = 0; u < 2; ++u) {
        const int c = lane + u * 32;
        const float zt = Gr[c]        + beff[c];
        const float it = Gr[64 + c]   + beff[64 + c];
        const float ft = Gr[128 + c]  + beff[128 + c];
        const float ot = Gr[192 + c]  + beff[192 + c];
        const float z = tanhf(zt), iv = __expf(it);
        const float f = sigmoidf(ft), o = sigmoidf(ot);
        const float c2 = f * cs[r * DD + c] + iv * z;
        const float n2 = f * ns[r * DD + c] + iv;
        cs[r * DD + c] = c2;  ns[r * DD + c] = n2;
        const float h = o * c2 / n2;
        hsv[u] = h;  s1 += h;  s2 += h * h;
      }
      s1 = wave_sum(s1); s2 = wave_sum(s2);
      const float mu = s1 * (1.f / 64.f);
      const float rstd = rsqrtf(s2 * (1.f / 64.f) - mu * mu + 1e-5f);
      float gi = 0.f, gf = 0.f;
#pragma unroll
      for (int u = 0; u < 2; ++u) {
        const int c = lane + u * 32;
        const float ho = (hsv[u] - mu) * rstd * vec[4 * DD + c] + vec[5 * DD + c];
        Hs[r * DD + c]  = (__bf16)hsv[u];
        Hso[r * DD + c] = (__bf16)ho;
        Cat[r * INC + c] = (__bf16)ho;
        gi += ho * vec[2 * DD + c];
        gf += ho * vec[3 * DD + c];
      }
      gi = wave_sum(gi); gf = wave_sum(gf);
      if (lane == 0) {
        sc[r]     = __expf(gi + biS);       // im
        sc[8 + r] = sigmoidf(gf + bfS);     // fm
      }
    }
    __syncthreads();

    // ---- GEMM2: G[16,256] = Hso @ [Wq|Wk*s|Wv|Wo] ----
    {
      v16bf ah[2];
#pragma unroll
      for (int kk = 0; kk < 2; ++kk) ah[kk] = load_a(Hso + kk * 32, DD, lane);
#pragma unroll
      for (int it = 0; it < 2; ++it) {
        const int n0 = (wv + it * 8) * 16;
        v8f acc = {0.f, 0.f, 0.f, 0.f, 0.f, 0.f, 0.f, 0.f};
#pragma unroll
        for (int kk = 0; kk < 2; ++kk)
          acc = WMMA_BF16(ah[kk], load_b(WqT + n0 * DD + kk * 32, DD, lane), acc);
        store_d(G + n0, FD, lane, acc);
      }
    }
    __syncthreads();

    // ---- n_m update + den (wave-per-row) ----
    {
      const int r = wv;
      const float im = sc[r], fm = sc[8 + r];
      float dd = 0.f;
#pragma unroll
      for (int u = 0; u < 2; ++u) {
        const int c = lane + u * 32;
        const float kv = G[r * FD + 64 + c], qv = G[r * FD + c];
        const float n2 = fm * nm[r * DD + c] + im * kv;
        nm[r * DD + c] = n2;
        dd += n2 * qv;
      }
      dd = wave_sum(dd);
      if (lane == 0) sc[16 + r] = fmaxf(fabsf(dd), 1.f);
    }
    // ---- fused C_m update + num = C_m q ----
    // C stored as float2 pairs, pair-swizzled: physical pair pp = (jp + i) & 31.
    // Lane i's b64 access hits word banks {2((jp+i)&31), +1}: across 32 consecutive
    // lanes this covers all 64 banks exactly once -> conflict-free ds_load/store_b64.
    {
#pragma unroll
      for (int u = 0; u < 2; ++u) {
        const int p = tid + u * NTHR;
        const int r = p >> 6, i = p & 63;
        const float fm = sc[8 + r];
        const float iv = sc[r] * G[r * FD + 128 + i];   // im * v_i
        float2* Crow2 = (float2*)(Cm + (r * DD + i) * DD);
        const float* Gr = G + r * FD;
        float acc = 0.f;
#pragma unroll 8
        for (int jp = 0; jp < DD / 2; ++jp) {
          const int pp = (jp + i) & 31;                 // pair-granular bank swizzle
          float2 c = Crow2[pp];
          c.x = fm * c.x + iv * Gr[64 + 2 * jp];
          c.y = fm * c.y + iv * Gr[64 + 2 * jp + 1];
          Crow2[pp] = c;
          acc += c.x * Gr[2 * jp] + c.y * Gr[2 * jp + 1];
        }
        numb[r * DD + i] = acc;
      }
    }
    __syncthreads();

    // ---- h_m = sigmoid(o)*num/den ; build [Hso|Hm] for fusion (wave-per-row) ----
    {
      const int r = wv;
      const float rdn = 1.f / sc[16 + r];
#pragma unroll
      for (int u = 0; u < 2; ++u) {
        const int c = lane + u * 32;
        const float om = sigmoidf(G[r * FD + 192 + c] + vec[0 * DD + c]);
        const float h = om * numb[r * DD + c] * rdn;
        hmb[r * DD + c] = h;
        Cat[r * INC + 64 + c] = (__bf16)h;
      }
    }

    // ---- fusion gate: h_f never feeds back -> only compute at the final step ----
    if (t == SS - 1) {
      __syncthreads();
      if (wv < 4) {                                     // N=64 -> 4 tiles
        v16bf ac[4];
#pragma unroll
        for (int kk = 0; kk < 4; ++kk) ac[kk] = load_a(Cat + kk * 32, INC, lane);
        const int n0 = wv * 16;
        v8f acc = {0.f, 0.f, 0.f, 0.f, 0.f, 0.f, 0.f, 0.f};
#pragma unroll
        for (int kk = 0; kk < 4; ++kk)
          acc = WMMA_BF16(ac[kk], load_b(WfT + n0 * INC + kk * 32, INC, lane), acc);
        store_d(G + n0, FD, lane, acc);
      }
      __syncthreads();
      const int r = wv;
#pragma unroll
      for (int u = 0; u < 2; ++u) {
        const int c = lane + u * 32;
        const float g = sigmoidf(G[r * FD + c] + vec[1 * DD + c]);
        const float hso = (float)Hso[r * DD + c];
        out[(size_t)(b0 + r) * DD + c] = g * hmb[r * DD + c] + (1.f - g) * hso;
      }
    }
  }
}

extern "C" void kernel_launch(void* const* d_in, const int* in_sizes, int n_in,
                              void* d_out, int out_size, void* d_ws, size_t ws_size,
                              hipStream_t stream) {
  (void)in_sizes; (void)n_in; (void)d_ws; (void)ws_size; (void)out_size;
  const float* x      = (const float*)d_in[0];
  const float* W_in   = (const float*)d_in[1];
  const float* b_in   = (const float*)d_in[2];
  const float* W_s    = (const float*)d_in[3];
  const float* R_s    = (const float*)d_in[4];
  const float* b_s    = (const float*)d_in[5];
  const float* ln_g   = (const float*)d_in[6];
  const float* ln_b   = (const float*)d_in[7];
  const float* Wq     = (const float*)d_in[8];
  const float* Wk     = (const float*)d_in[9];
  const float* Wv     = (const float*)d_in[10];
  const float* wi     = (const float*)d_in[11];
  const float* bi     = (const float*)d_in[12];
  const float* wf     = (const float*)d_in[13];
  const float* bf     = (const float*)d_in[14];
  const float* Wo     = (const float*)d_in[15];
  const float* bo     = (const float*)d_in[16];
  const float* W_fuse = (const float*)d_in[17];
  const float* b_fuse = (const float*)d_in[18];

  dim3 grid(128 / ROWS);   // 16 workgroups x 8 batch rows
  dim3 block(NTHR);
  hipLaunchKernelGGL(xlstm_fused_kernel, grid, block, SMEM_BYTES, stream,
                     x, W_in, b_in, W_s, R_s, b_s, ln_g, ln_b,
                     Wq, Wk, Wv, wi, bi, wf, bf, Wo, bo, W_fuse, b_fuse,
                     (float*)d_out);
}
